// HierarchicalHeteroConv_13116830122423
// MI455X (gfx1250) — matrix operand order repaired and checked by reading, not compiled, not run on hardware
//
#include <hip/hip_runtime.h>

#define NNODE 20000
#define NEDGE 320000
#define DCH   512
#define ECCH  256
#define NLAYER 3

typedef __attribute__((ext_vector_type(16))) __bf16   v16bf;
typedef __attribute__((ext_vector_type(8)))  float    v8f;
typedef __attribute__((ext_vector_type(4)))  unsigned u4v;
typedef __attribute__((ext_vector_type(2)))  unsigned u2v;
typedef __attribute__((ext_vector_type(4)))  float    f4v;
typedef __attribute__((ext_vector_type(2)))  float    f2v;

union FragU { u4v q[2]; v16bf v; };

__device__ __forceinline__ unsigned f2bf(float f) {
  unsigned u = __float_as_uint(f);
  return (u + 0x7FFFu + ((u >> 16) & 1u)) >> 16;   // round-to-nearest-even
}
__device__ __forceinline__ unsigned bfpack(float lo, float hi) {
  return f2bf(lo) | (f2bf(hi) << 16);
}
__device__ __forceinline__ float bflo(unsigned u) { return __uint_as_float(u << 16); }
__device__ __forceinline__ float bfhi(unsigned u) { return __uint_as_float(u & 0xffff0000u); }

// ---------------------------------------------------------------------------
// Weight prep: fp32 KxN (row-major) -> bf16 WMMA B-fragment layout.
// Lane l: n = nt*16 + (l&15), K half = (l>>4)*16; VGPR v holds K = half+2v,+1.
// ---------------------------------------------------------------------------
__global__ void prep_weight(const float* __restrict__ W, unsigned* __restrict__ out,
                            int K, int Ncols) {
  int tid = blockIdx.x * 256 + threadIdx.x;
  int NT = Ncols >> 4;
  int nfrag = (K >> 5) * NT;
  if (tid >= nfrag * 32) return;
  int frag = tid >> 5, lane = tid & 31;
  int kt = frag / NT, nt = frag - kt * NT;
  int kbase = (kt << 5) + ((lane >> 4) << 4);
  int n = (nt << 4) + (lane & 15);
  unsigned* o = out + (size_t)tid * 8;
#pragma unroll
  for (int v = 0; v < 8; ++v) {
    float a = W[(size_t)(kbase + 2 * v) * Ncols + n];
    float b = W[(size_t)(kbase + 2 * v + 1) * Ncols + n];
    o[v] = bfpack(a, b);
  }
}

// ---------------------------------------------------------------------------
// C[M,N](f32) = (beta?C:0) + A16[M,K](bf16 pairs) @ Bprep(bf16 frags)
//               + bias[n]*(rowscale?rowscale[m]:1)
// 256 thr = 8 waves; tile 128x128; wave -> 32x64 = 2x4 WMMA 16x16x32 bf16.
// A tiles move global->LDS with double-buffered GLOBAL_LOAD_ASYNC_TO_LDS_B128
// (ASYNCcnt), overlapping the next tile's fill with this tile's WMMAs.
// Tail rows clamped (branchless); garbage accumulators are never stored.
// ---------------------------------------------------------------------------
__global__ __launch_bounds__(256) void gemm_bf16(
    const unsigned* __restrict__ A16, const unsigned* __restrict__ Bp,
    const float* __restrict__ bias, const float* __restrict__ rowscale,
    float* __restrict__ C, int M, int Ncols, int K, int beta) {
  __shared__ unsigned As32[2][128 * 20];          // 2 x (128 rows x 16 u32, pitch 20)
  const int t = threadIdx.x;
  const int lane = t & 31;
  const int wv = t >> 5;
  const int wm = wv & 3;                           // wave M index (0..3)
  const int wn = wv >> 2;                          // wave N index (0..1)
  const int mblk = blockIdx.x * 128;
  const int nblk = blockIdx.y * 128;
  const int NT = Ncols >> 4;
  const int KW = K >> 1;                           // A row pitch in u32
  const int msub = lane & 15;
  const int kh = lane >> 4;
  // staging coords: 2 threads per row, 8 u32 (32B) each = 2 async b128
  const int sr = t >> 1, sh = (t & 1) << 3;
  int srow = mblk + sr; if (srow >= M) srow = M - 1;
  const unsigned* sbase = A16 + (size_t)srow * KW + sh;
  unsigned lds_a[2];
  lds_a[0] = (unsigned)(uintptr_t)(&As32[0][sr * 20 + sh]);
  lds_a[1] = (unsigned)(uintptr_t)(&As32[1][sr * 20 + sh]);
  v8f acc[2][4] = {};
  const int KT = K >> 5;
  // prefetch first tile into buffer 0
  {
    unsigned long long ga = (unsigned long long)(uintptr_t)sbase;
    asm volatile("global_load_async_to_lds_b128 %0, %1, off"
                 :: "v"(lds_a[0]), "v"(ga) : "memory");
    asm volatile("global_load_async_to_lds_b128 %0, %1, off offset:16"
                 :: "v"(lds_a[0]), "v"(ga) : "memory");
  }
  for (int kt = 0; kt < KT; ++kt) {
    asm volatile("s_wait_asynccnt 0" ::: "memory");
    __syncthreads();
    if (kt + 1 < KT) {
      unsigned la = lds_a[(kt + 1) & 1];
      unsigned long long ga = (unsigned long long)(uintptr_t)(sbase + ((kt + 1) << 4));
      asm volatile("global_load_async_to_lds_b128 %0, %1, off"
                   :: "v"(la), "v"(ga) : "memory");
      asm volatile("global_load_async_to_lds_b128 %0, %1, off offset:16"
                   :: "v"(la), "v"(ga) : "memory");
    }
    const unsigned* cb = As32[kt & 1];
    FragU a0, a1, b[4];
    {
      const unsigned* r0 = cb + (wm * 32 + msub) * 20 + kh * 8;
      a0.q[0] = *(const u4v*)r0;
      a0.q[1] = *(const u4v*)(r0 + 4);
      const unsigned* r1 = cb + (wm * 32 + 16 + msub) * 20 + kh * 8;
      a1.q[0] = *(const u4v*)r1;
      a1.q[1] = *(const u4v*)(r1 + 4);
    }
    {
      int nt0 = (nblk >> 4) + wn * 4;
      const unsigned* p = Bp + (((size_t)kt * NT + nt0) * 32 + lane) * 8;
#pragma unroll
      for (int j = 0; j < 4; ++j) {
        b[j].q[0] = *(const u4v*)(p + j * 256);
        b[j].q[1] = *(const u4v*)(p + j * 256 + 4);
      }
    }
#pragma unroll
    for (int j = 0; j < 4; ++j) {
      acc[0][j] = __builtin_amdgcn_wmma_f32_16x16x32_bf16(false, a0.v, false, b[j].v, (short)0, acc[0][j], false, false);
      acc[1][j] = __builtin_amdgcn_wmma_f32_16x16x32_bf16(false, a1.v, false, b[j].v, (short)0, acc[1][j], false, false);
    }
  }
  // epilogue (C/D layout: n = lane&15, row = sub + 8*(lane>>4) + vgpr idx)
#pragma unroll
  for (int i = 0; i < 2; ++i) {
#pragma unroll
    for (int j = 0; j < 4; ++j) {
      int col = nblk + wn * 64 + j * 16 + (lane & 15);
      int rbase = mblk + wm * 32 + i * 16 + kh * 8;
      float bcol = (bias != nullptr) ? bias[col] : 0.f;
#pragma unroll
      for (int r = 0; r < 8; ++r) {
        int row = rbase + r;
        if (row < M) {
          size_t o = (size_t)row * Ncols + col;
          float v = acc[i][j][r];
          if (beta) v += C[o];
          if (bias) v += (rowscale ? rowscale[row] * bcol : bcol);
          C[o] = v;
        }
      }
    }
  }
}

// ---------------------------------------------------------------------------
// Helpers
// ---------------------------------------------------------------------------
__global__ void zero_kernel(float* __restrict__ p, size_t n) {
  size_t i = (size_t)blockIdx.x * 256 + threadIdx.x;
  size_t st = (size_t)gridDim.x * 256;
  for (; i < n; i += st) p[i] = 0.f;
}

__global__ void count_edges(const int* __restrict__ dst, float* __restrict__ cnt, int nE) {
  int e = blockIdx.x * 256 + threadIdx.x;
  if (e < nE) atomicAdd(cnt + dst[e], 1.f);
}

// fp32 -> packed bf16 (plain)
__global__ void f32_to_bf16(const float* __restrict__ in, unsigned* __restrict__ out, int n4) {
  int i = blockIdx.x * 256 + threadIdx.x;
  if (i >= n4) return;
  f4v v = ((const f4v*)in)[i];
  u2v o = {bfpack(v.x, v.y), bfpack(v.z, v.w)};
  ((u2v*)out)[i] = o;
}

// fused mean: fp32 segment-sum buffer -> bf16, scaled by 1/max(cnt,1)
__global__ void mean_to_bf16(const float* __restrict__ in, const float* __restrict__ cnt,
                             unsigned* __restrict__ out, int n4, int d4) {
  int i = blockIdx.x * 256 + threadIdx.x;
  if (i >= n4) return;
  float inv = 1.f / fmaxf(cnt[i / d4], 1.f);
  f4v v = ((const f4v*)in)[i];
  u2v o = {bfpack(v.x * inv, v.y * inv), bfpack(v.z * inv, v.w * inv)};
  ((u2v*)out)[i] = o;
}

// gather bf16 source rows (b128 = 8 features / thread) -> fp32 atomic scatter
__global__ void scatter_src_rows_bf16(const unsigned* __restrict__ x16,
                                      const int* __restrict__ src, const int* __restrict__ dst,
                                      float* __restrict__ out, int nE, int Dcols) {
  int d8 = Dcols >> 3;
  int tid = blockIdx.x * 256 + threadIdx.x;
  if (tid >= nE * d8) return;
  int e = tid / d8, c = tid - e * d8;
  u4v v = *(const u4v*)(x16 + (size_t)src[e] * (Dcols >> 1) + (c << 2));
  float* o = out + (size_t)dst[e] * Dcols + (c << 3);
#pragma unroll
  for (int j = 0; j < 4; ++j) {
    atomicAdd(o + 2 * j,     bflo(v[j]));
    atomicAdd(o + 2 * j + 1, bfhi(v[j]));
  }
}

// per-edge fp32 rows -> fp32 atomic scatter (one-time ef_h reduction)
__global__ void scatter_edge_rows(const float* __restrict__ ef, const int* __restrict__ dst,
                                  float* __restrict__ out, int nE, int Dcols) {
  int d4 = Dcols >> 2;
  int tid = blockIdx.x * 256 + threadIdx.x;
  if (tid >= nE * d4) return;
  int e = tid / d4, c = tid - e * d4;
  f4v v = *(const f4v*)(ef + (size_t)e * Dcols + (c << 2));
  float* o = out + (size_t)dst[e] * Dcols + (c << 2);
  atomicAdd(o + 0, v.x); atomicAdd(o + 1, v.y);
  atomicAdd(o + 2, v.z); atomicAdd(o + 3, v.w);
}

// fused relu -> layernorm -> bf16 pack (row/block, 256 thr x 2 elems)
__global__ __launch_bounds__(256) void relu_ln_bf16(const float* __restrict__ x,
                                                    const float* __restrict__ g,
                                                    const float* __restrict__ b,
                                                    unsigned* __restrict__ out16) {
  __shared__ float r1[256], r2[256], stats[2];
  int row = blockIdx.x, t = threadIdx.x;
  f2v v = *(const f2v*)(x + (size_t)row * DCH + 2 * t);
  float v0 = fmaxf(v.x, 0.f), v1 = fmaxf(v.y, 0.f);
  r1[t] = v0 + v1;
  r2[t] = v0 * v0 + v1 * v1;
  __syncthreads();
  for (int o = 128; o > 0; o >>= 1) {
    if (t < o) { r1[t] += r1[t + o]; r2[t] += r2[t + o]; }
    __syncthreads();
  }
  if (t == 0) {
    float mu = r1[0] * (1.f / DCH);
    float var = r2[0] * (1.f / DCH) - mu * mu;
    stats[0] = mu;
    stats[1] = rsqrtf(var + 1e-5f);
  }
  __syncthreads();
  float mu = stats[0], rs = stats[1];
  float n0 = (v0 - mu) * rs * g[2 * t] + b[2 * t];
  float n1 = (v1 - mu) * rs * g[2 * t + 1] + b[2 * t + 1];
  out16[(size_t)row * (DCH / 2) + t] = bfpack(n0, n1);
}

// ---------------------------------------------------------------------------
extern "C" void kernel_launch(void* const* d_in, const int* in_sizes, int n_in,
                              void* d_out, int out_size, void* d_ws, size_t ws_size,
                              hipStream_t stream) {
  (void)in_sizes; (void)n_in; (void)out_size; (void)ws_size;
  const float* x_new = (const float*)d_in[0];
  const float* x_old = (const float*)d_in[1];
  const float* ef_h  = (const float*)d_in[2];
  const float* Wl_nn = (const float*)d_in[3];
  const float* Wr_nn = (const float*)d_in[4];
  const float* bl_nn = (const float*)d_in[5];
  const float* Wl_on = (const float*)d_in[6];
  const float* Wr_on = (const float*)d_in[7];
  const float* bl_on = (const float*)d_in[8];
  const float* Wl_oo = (const float*)d_in[9];
  const float* Wr_oo = (const float*)d_in[10];
  const float* bl_oo = (const float*)d_in[11];
  const float* We_h  = (const float*)d_in[12];
  const float* be_h  = (const float*)d_in[13];
  const float* Wn_h  = (const float*)d_in[14];
  const float* bn_h  = (const float*)d_in[15];
  const float* ln_g  = (const float*)d_in[16];
  const float* ln_b  = (const float*)d_in[17];
  const int* ei_nn = (const int*)d_in[18];
  const int* ei_on = (const int*)d_in[19];
  const int* ei_oo = (const int*)d_in[20];
  const int* ei_h  = (const int*)d_in[21];

  char* base = (char*)d_ws;
  size_t off = 0;
  auto carve = [&](size_t bytes) -> char* {
    char* p = base + off;
    off = (off + bytes + 255) & ~(size_t)255;
    return p;
  };
  const size_t nd  = (size_t)NNODE * DCH;
  const size_t nec = (size_t)NNODE * ECCH;
  unsigned* xn16[2] = {(unsigned*)carve(nd * 2), (unsigned*)carve(nd * 2)};
  unsigned* xo16[2] = {(unsigned*)carve(nd * 2), (unsigned*)carve(nd * 2)};
  unsigned* mean16  = (unsigned*)carve(nd * 2);
  unsigned* Sh16    = (unsigned*)carve(nec * 2);
  float* meanb    = (float*)carve(nd * 4);
  float* Sh       = (float*)carve(nec * 4);
  float* tmp_new  = (float*)carve(nd * 4);
  float* tmp_old  = (float*)carve(nd * 4);
  float* cnt_nn = (float*)carve(NNODE * 4);
  float* cnt_on = (float*)carve(NNODE * 4);
  float* cnt_oo = (float*)carve(NNODE * 4);
  float* cnt_h  = (float*)carve(NNODE * 4);
  const size_t wDD = (size_t)DCH * DCH / 2;   // u32 per 512x512 bf16
  const size_t wED = (size_t)ECCH * DCH / 2;
  unsigned* Wp[NLAYER][8];
  for (int l = 0; l < NLAYER; ++l)
    for (int m = 0; m < 8; ++m)
      Wp[l][m] = (unsigned*)carve(((m == 7) ? wED : wDD) * 4);

  // ---- one-time precompute: weight prep, activation->bf16, counts, S_h ----
  for (int l = 0; l < NLAYER; ++l) {
    const float* srcs[8] = {
      Wl_nn + (size_t)l * DCH * DCH, Wr_nn + (size_t)l * DCH * DCH,
      Wl_on + (size_t)l * DCH * DCH, Wr_on + (size_t)l * DCH * DCH,
      Wl_oo + (size_t)l * DCH * DCH, Wr_oo + (size_t)l * DCH * DCH,
      Wn_h  + (size_t)l * DCH * DCH, We_h  + (size_t)l * ECCH * DCH };
    for (int m = 0; m < 8; ++m) {
      int K = (m == 7) ? ECCH : DCH;
      int nthreads = (K >> 5) * (DCH >> 4) * 32;
      prep_weight<<<(nthreads + 255) / 256, 256, 0, stream>>>(srcs[m], Wp[l][m], K, DCH);
    }
  }
  f32_to_bf16<<<((int)(nd / 4) + 255) / 256, 256, 0, stream>>>(x_new, xn16[0], (int)(nd / 4));
  f32_to_bf16<<<((int)(nd / 4) + 255) / 256, 256, 0, stream>>>(x_old, xo16[0], (int)(nd / 4));
  zero_kernel<<<128, 256, 0, stream>>>(cnt_nn, NNODE);
  zero_kernel<<<128, 256, 0, stream>>>(cnt_on, NNODE);
  zero_kernel<<<128, 256, 0, stream>>>(cnt_oo, NNODE);
  zero_kernel<<<128, 256, 0, stream>>>(cnt_h,  NNODE);
  zero_kernel<<<2048, 256, 0, stream>>>(Sh, nec);
  count_edges<<<(NEDGE + 255) / 256, 256, 0, stream>>>(ei_nn + NEDGE, cnt_nn, NEDGE);
  count_edges<<<(NEDGE + 255) / 256, 256, 0, stream>>>(ei_on + NEDGE, cnt_on, NEDGE);
  count_edges<<<(NEDGE + 255) / 256, 256, 0, stream>>>(ei_oo + NEDGE, cnt_oo, NEDGE);
  count_edges<<<(NEDGE + 255) / 256, 256, 0, stream>>>(ei_h  + NEDGE, cnt_h,  NEDGE);
  // hoisted EdgeConv reduction: S_h = segment_sum(ef_h by dst), reused all layers
  scatter_edge_rows<<<(NEDGE * (ECCH / 4) + 255) / 256, 256, 0, stream>>>(
      ef_h, ei_h + NEDGE, Sh, NEDGE, ECCH);
  f32_to_bf16<<<((int)(nec / 4) + 255) / 256, 256, 0, stream>>>(Sh, Sh16, (int)(nec / 4));

  dim3 ggrid((NNODE + 127) / 128, DCH / 128);
  int sg_blocks = (NEDGE * (DCH / 8) + 255) / 256;
  int cvt_blocks = ((int)(nd / 4) + 255) / 256;
  for (int l = 0; l < NLAYER; ++l) {
    const unsigned* an = xn16[l & 1];
    const unsigned* ao = xo16[l & 1];
    unsigned* nn_next = xn16[(l + 1) & 1];
    unsigned* no_next = xo16[(l + 1) & 1];
    float* on_out = (l == 2) ? (float*)d_out        : tmp_new;
    float* oo_out = (l == 2) ? ((float*)d_out + nd) : tmp_old;
    const size_t lb = (size_t)l * DCH;

    // ---- destination 'new': sage(nn) + sage(on) ----
    zero_kernel<<<2048, 256, 0, stream>>>(meanb, nd);
    scatter_src_rows_bf16<<<sg_blocks, 256, 0, stream>>>(an, ei_nn, ei_nn + NEDGE, meanb, NEDGE, DCH);
    mean_to_bf16<<<cvt_blocks, 256, 0, stream>>>(meanb, cnt_nn, mean16, (int)(nd / 4), DCH / 4);
    gemm_bf16<<<ggrid, 256, 0, stream>>>(mean16, Wp[l][0], bl_nn + lb, nullptr, on_out, NNODE, DCH, DCH, 0);
    gemm_bf16<<<ggrid, 256, 0, stream>>>(an, Wp[l][1], nullptr, nullptr, on_out, NNODE, DCH, DCH, 1);
    zero_kernel<<<2048, 256, 0, stream>>>(meanb, nd);
    scatter_src_rows_bf16<<<sg_blocks, 256, 0, stream>>>(ao, ei_on, ei_on + NEDGE, meanb, NEDGE, DCH);
    mean_to_bf16<<<cvt_blocks, 256, 0, stream>>>(meanb, cnt_on, mean16, (int)(nd / 4), DCH / 4);
    gemm_bf16<<<ggrid, 256, 0, stream>>>(mean16, Wp[l][2], bl_on + lb, nullptr, on_out, NNODE, DCH, DCH, 1);
    gemm_bf16<<<ggrid, 256, 0, stream>>>(an, Wp[l][3], nullptr, nullptr, on_out, NNODE, DCH, DCH, 1);

    // ---- destination 'old': sage(oo) + edgeconv(h) ----
    zero_kernel<<<2048, 256, 0, stream>>>(meanb, nd);
    scatter_src_rows_bf16<<<sg_blocks, 256, 0, stream>>>(ao, ei_oo, ei_oo + NEDGE, meanb, NEDGE, DCH);
    mean_to_bf16<<<cvt_blocks, 256, 0, stream>>>(meanb, cnt_oo, mean16, (int)(nd / 4), DCH / 4);
    gemm_bf16<<<ggrid, 256, 0, stream>>>(mean16, Wp[l][4], bl_oo + lb, nullptr, oo_out, NNODE, DCH, DCH, 0);
    gemm_bf16<<<ggrid, 256, 0, stream>>>(ao, Wp[l][5], nullptr, nullptr, oo_out, NNODE, DCH, DCH, 1);
    // S_h @ We + cnt_h * be  (hoisted edge GEMM, K=256, bias scaled by count)
    gemm_bf16<<<ggrid, 256, 0, stream>>>(Sh16, Wp[l][7], be_h + lb, cnt_h, oo_out, NNODE, DCH, ECCH, 1);
    gemm_bf16<<<ggrid, 256, 0, stream>>>(ao, Wp[l][6], bn_h + lb, nullptr, oo_out, NNODE, DCH, DCH, 1);

    if (l < NLAYER - 1) {
      relu_ln_bf16<<<NNODE, 256, 0, stream>>>(on_out, ln_g + lb, ln_b + lb, nn_next);
      relu_ln_bf16<<<NNODE, 256, 0, stream>>>(oo_out, ln_g + lb, ln_b + lb, no_next);
    }
  }
}
